// DocREModel_Triangle_79809082294511
// MI455X (gfx1250) — matrix operand order repaired and checked by reading, not compiled, not run on hardware
//
#include <hip/hip_runtime.h>

// ---------------------------------------------------------------------------
// Types
// ---------------------------------------------------------------------------
typedef __bf16 bf16;
typedef __attribute__((ext_vector_type(8)))  __bf16 bf16x8;
typedef __attribute__((ext_vector_type(16))) __bf16 bf16x16;
typedef __attribute__((ext_vector_type(8)))  float  f32x8;

#define DEVI __device__ __forceinline__

// ---------------------------------------------------------------------------
// Problem constants (from the reference)
// ---------------------------------------------------------------------------
constexpr int Bc = 2, Lc = 512, Hc = 768, NHc = 12, NEc = 24, Mc = 4;
constexpr int Ec = 768, BSc = 64, Cc = 97;
constexpr int PAIRS = NEc * NEc;          // 576
constexpr int ROWS  = Bc * PAIRS;         // 1152
constexpr int KBIG  = Ec * BSc;           // 49152 (bilinear reduction dim)

// ---------------------------------------------------------------------------
// gfx1250 async global->LDS via inline asm (ISA: GLOBAL_LOAD_ASYNC_TO_LDS_B128,
// VGLOBAL encoding "GV": vdst = LDS byte address VGPR, vaddr = 64-bit global
// address VGPR pair, no SADDR -> "off"). Tracked by ASYNCcnt.
// ---------------------------------------------------------------------------
#if defined(__gfx1250__)
#define HAVE_ASYNC_LDS 1
#else
#define HAVE_ASYNC_LDS 0
#endif

#if HAVE_ASYNC_LDS
#define WAIT_ASYNC(N) asm volatile("s_wait_asynccnt %0" ::"i"(N) : "memory")
#else
#define WAIT_ASYNC(N) ((void)0)
#endif

DEVI void async_load16(const void* g, void* l) {
#if HAVE_ASYNC_LDS
  const unsigned lds_addr = (unsigned)(unsigned long long)l;
  asm volatile("global_load_async_to_lds_b128 %0, %1, off"
               :
               : "v"(lds_addr), "v"((unsigned long long)g)
               : "memory");
#else
  *(uint4*)l = *(const uint4*)g;
#endif
}

// ---------------------------------------------------------------------------
// WMMA helpers (V_WMMA_F32_16X16X32_BF16, wave32)
//   A (16x32, bf16): lane l -> row M=l&15, half h=l>>4;
//       elems 0..7  -> K = 8h + 0..7
//       elems 8..15 -> K = 16 + 8h + 0..7
//   B (32x16, bf16): lane l -> col N=l&15, half h=l>>4; elems 0..7 -> K = 8h + 0..7, 8..15 -> K = 16 + 8h + 0..7
//   C/D (16x16,f32): lane l -> col N=l&15; vgpr v -> row M = v + 8*(l>>4)
// ---------------------------------------------------------------------------
DEVI f32x8 wmma_bf16(bf16x16 a, bf16x16 b, f32x8 c) {
  return __builtin_amdgcn_wmma_f32_16x16x32_bf16(false, a, false, b, (short)0, c,
                                                 false, false);
}

DEVI bf16x16 join8(bf16x8 lo, bf16x8 hi) {
  return __builtin_shufflevector(lo, hi, 0, 1, 2, 3, 4, 5, 6, 7, 8, 9, 10, 11,
                                 12, 13, 14, 15);
}

// A-fragment from a row-major bf16 matrix (row stride = pitch, K contiguous)
DEVI bf16x16 load_a_frag(const bf16* A, int row0, int pitch, int k0, int lane) {
  const int m = lane & 15, h = lane >> 4;
  const bf16* p = A + (size_t)(row0 + m) * pitch + k0 + h * 8;
  bf16x8 lo = *(const bf16x8*)p;
  bf16x8 hi = *(const bf16x8*)(p + 16);
  return join8(lo, hi);
}

// B-fragment from a TRANSPOSED bf16 matrix BT[N][K] (K contiguous per column)
DEVI bf16x16 load_bT_frag(const bf16* BT, int col0, int pitch, int k0, int lane) {
  const int n = lane & 15, h = lane >> 4;
  const bf16* p = BT + (size_t)(col0 + n) * pitch + k0 + h * 8;
  bf16x8 lo = *(const bf16x8*)p;
  bf16x8 hi = *(const bf16x8*)(p + 16);
  return join8(lo, hi);
}

// ---------------------------------------------------------------------------
// K0: tiled f32 -> bf16 transpose:  dst[c][r] = bf16(src[row0+r][c])
// ---------------------------------------------------------------------------
__global__ __launch_bounds__(256) void transpose_f32_bf16_kernel(
    const float* __restrict__ src, bf16* __restrict__ dst, int R, int C,
    int src_pitch, int dst_pitch, long long src_bstride, long long dst_bstride,
    int row0) {
  __shared__ float tile[32][33];
  src += (size_t)blockIdx.z * src_bstride + (size_t)row0 * src_pitch;
  dst += (size_t)blockIdx.z * dst_bstride;
  const int r0 = blockIdx.x * 32, c0 = blockIdx.y * 32;
  for (int i = threadIdx.y; i < 32; i += 8) {
    int r = r0 + i, c = c0 + threadIdx.x;
    tile[i][threadIdx.x] = (r < R && c < C) ? src[(size_t)r * src_pitch + c] : 0.f;
  }
  __syncthreads();
  for (int i = threadIdx.y; i < 32; i += 8) {
    int c = c0 + i, r = r0 + threadIdx.x;
    if (c < C && r < R) dst[(size_t)c * dst_pitch + r] = (bf16)tile[threadIdx.x][i];
  }
}

// ---------------------------------------------------------------------------
// K1: entity_embs[b,e,h] = logsumexp_m seq[b, pos[b,e,m], h]
// ---------------------------------------------------------------------------
__global__ void entity_embs_kernel(const float* __restrict__ seq,
                                   const int* __restrict__ mpos,
                                   float* __restrict__ ee) {
  const int idx = blockIdx.x * blockDim.x + threadIdx.x;
  if (idx >= Bc * NEc * Hc) return;
  const int hid = idx % Hc;
  const int e = (idx / Hc) % NEc;
  const int b = idx / (Hc * NEc);
  const int* mp = mpos + (b * NEc + e) * Mc;
  float x[Mc], mx = -3.0e38f;
#pragma unroll
  for (int m = 0; m < Mc; ++m) {
    x[m] = seq[((size_t)b * Lc + mp[m]) * Hc + hid];
    mx = fmaxf(mx, x[m]);
  }
  float s = 0.f;
#pragma unroll
  for (int m = 0; m < Mc; ++m) s += __expf(x[m] - mx);
  ee[idx] = mx + __logf(s);
}

// ---------------------------------------------------------------------------
// K2: e_att[b,e,h,l] = mean_m attention[b,h,pos[b,e,m],l]
// ---------------------------------------------------------------------------
__global__ void e_att_kernel(const float* __restrict__ att,
                             const int* __restrict__ mpos,
                             float* __restrict__ eatt) {
  const int idx = blockIdx.x * blockDim.x + threadIdx.x;
  if (idx >= Bc * NEc * NHc * Lc) return;
  const int l = idx & (Lc - 1);
  const int hh = (idx / Lc) % NHc;
  const int e = (idx / (Lc * NHc)) % NEc;
  const int b = idx / (Lc * NHc * NEc);
  const int* mp = mpos + (b * NEc + e) * Mc;
  float s = 0.f;
#pragma unroll
  for (int m = 0; m < Mc; ++m)
    s += att[(((size_t)b * NHc + hh) * Lc + mp[m]) * Lc + l];
  eatt[idx] = 0.25f * s;
}

// ---------------------------------------------------------------------------
// K3: ht_att[b,i,j,l] = relu(sum_h e_att[b,i,h,l]*e_att[b,j,h,l]), then
//     normalize over l, write bf16 A-matrix htA[b][i*NE+j][l].
//     One block per (b,i,j), 256 threads x 2 l-values.
// ---------------------------------------------------------------------------
__global__ __launch_bounds__(256) void ht_att_kernel(const float* __restrict__ eatt,
                                                     bf16* __restrict__ htA) {
  const int t = blockIdx.x;
  const int j = t % NEc;
  const int i = (t / NEc) % NEc;
  const int b = t / (NEc * NEc);
  const float* ai = eatt + ((size_t)(b * NEc + i)) * NHc * Lc;
  const float* aj = eatt + ((size_t)(b * NEc + j)) * NHc * Lc;

  float v[2], lsum = 0.f;
#pragma unroll
  for (int u = 0; u < 2; ++u) {
    const int l = threadIdx.x + u * 256;
    float s = 0.f;
#pragma unroll
    for (int hh = 0; hh < NHc; ++hh) s += ai[hh * Lc + l] * aj[hh * Lc + l];
    s = fmaxf(s, 0.f);
    v[u] = s;
    lsum += s;
  }
  __shared__ float red[256];
  red[threadIdx.x] = lsum;
  __syncthreads();
  for (int s = 128; s > 0; s >>= 1) {
    if (threadIdx.x < s) red[threadIdx.x] += red[threadIdx.x + s];
    __syncthreads();
  }
  const float inv = 1.f / (red[0] + 1e-10f);
  bf16* orow = htA + ((size_t)b * PAIRS + i * NEc + j) * Lc;
#pragma unroll
  for (int u = 0; u < 2; ++u) orow[threadIdx.x + u * 256] = (bf16)(v[u] * inv);
}

// ---------------------------------------------------------------------------
// K4: part[b,i,e] = entity_embs[b,i,:] @ W[:H,:]   (W row-major [2H,E])
// ---------------------------------------------------------------------------
__global__ __launch_bounds__(256) void part_kernel(const float* __restrict__ ee,
                                                   const float* __restrict__ W,
                                                   float* __restrict__ part) {
  __shared__ float row[Hc];
  const int bi = blockIdx.x;  // 0..B*NE-1
  const float* er = ee + (size_t)bi * Hc;
  for (int k = threadIdx.x; k < Hc; k += 256) row[k] = er[k];
  __syncthreads();
  for (int e = threadIdx.x; e < Ec; e += 256) {
    float acc = 0.f;
    for (int k = 0; k < Hc; ++k) acc += row[k] * W[(size_t)k * Ec + e];
    part[(size_t)bi * Ec + e] = acc;
  }
}

// ---------------------------------------------------------------------------
// K5: rs GEMM (per batch): [576,512](htA bf16) x [512,768](seqT bf16) -> rs bf16
//     wave-per-16x16-tile, f32 accumulate. 3456 tiles / 8 waves = 432 blocks.
// ---------------------------------------------------------------------------
__global__ __launch_bounds__(256) void rs_gemm_kernel(const bf16* __restrict__ htA,
                                                      const bf16* __restrict__ seqT,
                                                      bf16* __restrict__ rs) {
  const int wave = threadIdx.x >> 5, lane = threadIdx.x & 31;
  const int tile = blockIdx.x * 8 + wave;
  const int b = tile / (36 * 48);
  const int rem = tile % (36 * 48);
  const int mt = rem / 48, nt = rem % 48;
  const bf16* A = htA + (size_t)b * PAIRS * Lc;
  const bf16* BT = seqT + (size_t)b * Hc * Lc;
  f32x8 acc = {0, 0, 0, 0, 0, 0, 0, 0};
  for (int kt = 0; kt < Lc / 32; ++kt)
    acc = wmma_bf16(load_a_frag(A, mt * 16, Lc, kt * 32, lane),
                    load_bT_frag(BT, nt * 16, Lc, kt * 32, lane), acc);
  const int n = lane & 15, h = lane >> 4;
  bf16* op = rs + ((size_t)b * PAIRS + mt * 16 + 8 * h) * Hc + nt * 16 + n;
#pragma unroll
  for (int v = 0; v < 8; ++v) op[(size_t)v * Hc] = (bf16)acc[v];
}

// ---------------------------------------------------------------------------
// K6: extractor GEMM: out = tanh(rs @ WT^T + part + bias), bf16 out.
//     headMode=1 -> part indexed by i, 0 -> by j.
// ---------------------------------------------------------------------------
__global__ __launch_bounds__(256) void extract_gemm_kernel(
    const bf16* __restrict__ rs, const bf16* __restrict__ WT,
    const float* __restrict__ part, const float* __restrict__ bias,
    bf16* __restrict__ out, int headMode) {
  const int wave = threadIdx.x >> 5, lane = threadIdx.x & 31;
  const int tile = blockIdx.x * 8 + wave;  // 72*48 tiles
  const int mt = tile / 48, nt = tile % 48;
  f32x8 acc = {0, 0, 0, 0, 0, 0, 0, 0};
  for (int kt = 0; kt < Hc / 32; ++kt)
    acc = wmma_bf16(load_a_frag(rs, mt * 16, Hc, kt * 32, lane),
                    load_bT_frag(WT, nt * 16, Hc, kt * 32, lane), acc);
  const int n = lane & 15, h = lane >> 4;
  const int col = nt * 16 + n;
  const float bia = bias[col];
#pragma unroll
  for (int v = 0; v < 8; ++v) {
    const int r = mt * 16 + v + 8 * h;
    const int b = r / PAIRS, ij = r % PAIRS;
    const int idx = headMode ? (ij / NEc) : (ij % NEc);
    const float p = part[((size_t)b * NEc + idx) * Ec + col];
    out[(size_t)r * Ec + col] = (bf16)tanhf(acc[v] + p + bia);
  }
}

// ---------------------------------------------------------------------------
// K7: bilinear projection (the 87-GFLOP GEMM):
//     feature[1152,768] = bl[1152,49152] @ Wp, with bl generated on the fly:
//     bl[r, k*4096 + s*64 + t] = hs[r, 64k+s] * ts[r, 64k+t]
//     WpT bf16 [768][49152] tiles async-staged to LDS, double-buffered.
//     Block: 32 rows x 256 cols; wave = (row-tile, 64-col group), 4 WMMA tiles.
// ---------------------------------------------------------------------------
__global__ __launch_bounds__(256) void bilinear_proj_kernel(
    const bf16* __restrict__ hs, const bf16* __restrict__ ts,
    const bf16* __restrict__ WpT, float* __restrict__ feat) {
  constexpr int NCH = KBIG / 32;  // 1536 chunks of 32 K-rows
  constexpr int LP = 40;          // padded LDS pitch (bf16) -> conflict-free
  __shared__ bf16 sB[2][256 * LP];

  const int tid = threadIdx.x;
  const int wave = tid >> 5, lane = tid & 31;
  const int rbase = blockIdx.x * 32 + (wave & 1) * 16;
  const int cg = wave >> 1;             // 0..3 -> which 64-col group
  const int cblk = blockIdx.y * 256;    // block's column base
  const int m = lane & 15, h = lane >> 4;
  const int row = rbase + m;

  const bf16* hsrow = hs + (size_t)row * Ec;
  const bf16* tsrow = ts + (size_t)row * Ec;
  const bf16* gsrc_row = WpT + (size_t)(cblk + tid) * KBIG;  // tid = local col

  f32x8 acc[4];
  const f32x8 z = {0, 0, 0, 0, 0, 0, 0, 0};
#pragma unroll
  for (int ct = 0; ct < 4; ++ct) acc[ct] = z;

  auto stage = [&](int buf, int n) {
    const bf16* src = gsrc_row + n * 32;
    bf16* dst = &sB[buf][tid * LP];
#pragma unroll
    for (int jj = 0; jj < 4; ++jj) async_load16(src + jj * 8, dst + jj * 8);
  };

  stage(0, 0);

  for (int n = 0; n < NCH; ++n) {
    const int cur = n & 1;
    __syncthreads();  // everyone done reading buf[cur^1]
    if (n + 1 < NCH) {
      stage(cur ^ 1, n + 1);
      WAIT_ASYNC(4);  // oldest 4 (buf[cur]) complete, newest 4 in flight
    } else {
      WAIT_ASYNC(0);
    }
    __syncthreads();  // buf[cur] visible to all waves

    const int k = n >> 7, s = (n >> 1) & 63, half = n & 1;

    // --- generate A fragment: a[m, kk] = hs[row, 64k+s] * ts[row, 64k+32h'+kk]
    const float hv = (float)hsrow[k * 64 + s];
    const bf16x8* tp = (const bf16x8*)(tsrow + k * 64 + half * 32 + h * 8);
    const bf16x8 tlo = tp[0];
    const bf16x8 thi = tp[2];  // +16 elements
    bf16x16 a;
#pragma unroll
    for (int i = 0; i < 8; ++i) {
      a[i] = (bf16)(hv * (float)tlo[i]);
      a[i + 8] = (bf16)(hv * (float)thi[i]);
    }

    // --- 4 col tiles from LDS (table-order K reads, padded pitch)
    const bf16* bbase = &sB[cur][(cg * 64 + m) * LP + h * 8];
#pragma unroll
    for (int ct = 0; ct < 4; ++ct) {
      const bf16* bp = bbase + ct * 16 * LP;
      bf16x8 blo = *(const bf16x8*)bp;
      bf16x8 bhi = *(const bf16x8*)(bp + 16);
      acc[ct] = wmma_bf16(a, join8(blo, bhi), acc[ct]);
    }
  }

  // epilogue: C layout -> feature f32
#pragma unroll
  for (int ct = 0; ct < 4; ++ct) {
    float* op = feat + (size_t)(rbase + 8 * h) * Ec + cblk + cg * 64 + ct * 16 + m;
#pragma unroll
    for (int v = 0; v < 8; ++v) op[(size_t)v * Ec] = acc[ct][v];
  }
}

// ---------------------------------------------------------------------------
// K8: classifier: out[r,c] = (i!=j) ? feature[r,:] @ Wc[:,c] + bc[c] : 0
// ---------------------------------------------------------------------------
__global__ __launch_bounds__(128) void classifier_kernel(
    const float* __restrict__ feat, const float* __restrict__ Wc,
    const float* __restrict__ bc, float* __restrict__ out) {
  const int r = blockIdx.x;
  __shared__ float frow[Hc];
  for (int k = threadIdx.x; k < Hc; k += 128) frow[k] = feat[(size_t)r * Hc + k];
  __syncthreads();
  const int ij = r % PAIRS;
  const int i = ij / NEc, j = ij % NEc;
  const int c = threadIdx.x;
  if (c < Cc) {
    float accv = 0.f;
    if (i != j) {
      accv = bc[c];
      for (int k = 0; k < Hc; ++k) accv += frow[k] * Wc[(size_t)k * Cc + c];
    }
    out[(size_t)r * Cc + c] = accv;
  }
}

// ---------------------------------------------------------------------------
// Host launcher
// ---------------------------------------------------------------------------
extern "C" void kernel_launch(void* const* d_in, const int* in_sizes, int n_in,
                              void* d_out, int out_size, void* d_ws,
                              size_t ws_size, hipStream_t stream) {
  (void)in_sizes; (void)n_in; (void)out_size; (void)ws_size;
  const float* seq = (const float*)d_in[0];   // [B,L,H]
  const float* att = (const float*)d_in[1];   // [B,NH,L,L]
  const float* Wh = (const float*)d_in[2];    // [2H,E]
  const float* bh = (const float*)d_in[3];    // [E]
  const float* Wt = (const float*)d_in[4];    // [2H,E]
  const float* bt = (const float*)d_in[5];    // [E]
  const float* Wp = (const float*)d_in[6];    // [E*BS,H]
  const float* Wc = (const float*)d_in[7];    // [H,C]
  const float* bc = (const float*)d_in[8];    // [C]
  const int* mpos = (const int*)d_in[9];      // [B,NE,M]
  float* out = (float*)d_out;

  // workspace layout (~87 MB)
  char* ws = (char*)d_ws;
  size_t off = 0;
  auto alloc = [&](size_t bytes) -> void* {
    void* p = ws + off;
    off = (off + bytes + 255) & ~(size_t)255;
    return p;
  };
  bf16* WpT  = (bf16*)alloc((size_t)Hc * KBIG * 2);          // [768][49152]
  bf16* seqT = (bf16*)alloc((size_t)Bc * Hc * Lc * 2);       // [B][768][512]
  bf16* WhT  = (bf16*)alloc((size_t)Ec * Hc * 2);            // [E][H]
  bf16* WtT  = (bf16*)alloc((size_t)Ec * Hc * 2);
  float* ee  = (float*)alloc((size_t)Bc * NEc * Hc * 4);
  float* eatt = (float*)alloc((size_t)Bc * NEc * NHc * Lc * 4);
  bf16* htA  = (bf16*)alloc((size_t)Bc * PAIRS * Lc * 2);
  bf16* rs   = (bf16*)alloc((size_t)Bc * PAIRS * Hc * 2);
  float* hpart = (float*)alloc((size_t)Bc * NEc * Ec * 4);
  float* tpart = (float*)alloc((size_t)Bc * NEc * Ec * 4);
  bf16* hsE  = (bf16*)alloc((size_t)ROWS * Ec * 2);
  bf16* tsE  = (bf16*)alloc((size_t)ROWS * Ec * 2);
  float* feat = (float*)alloc((size_t)ROWS * Hc * 4);

  dim3 tb(32, 8);
  // Wp: [49152,768] -> WpT bf16 [768][49152]
  transpose_f32_bf16_kernel<<<dim3(KBIG / 32, Hc / 32, 1), tb, 0, stream>>>(
      Wp, WpT, KBIG, Hc, Hc, KBIG, 0, 0, 0);
  // Wh[H:2H] -> WhT [E][H], Wt likewise
  transpose_f32_bf16_kernel<<<dim3(Hc / 32, Ec / 32, 1), tb, 0, stream>>>(
      Wh, WhT, Hc, Ec, Ec, Hc, 0, 0, Hc);
  transpose_f32_bf16_kernel<<<dim3(Hc / 32, Ec / 32, 1), tb, 0, stream>>>(
      Wt, WtT, Hc, Ec, Ec, Hc, 0, 0, Hc);
  // seq [B,512,768] -> seqT [B][768][512]
  transpose_f32_bf16_kernel<<<dim3(Lc / 32, Hc / 32, Bc), tb, 0, stream>>>(
      seq, seqT, Lc, Hc, Hc, Lc, (long long)Lc * Hc, (long long)Hc * Lc, 0);

  entity_embs_kernel<<<(Bc * NEc * Hc + 255) / 256, 256, 0, stream>>>(seq, mpos, ee);
  e_att_kernel<<<(Bc * NEc * NHc * Lc + 255) / 256, 256, 0, stream>>>(att, mpos, eatt);
  ht_att_kernel<<<Bc * PAIRS, 256, 0, stream>>>(eatt, htA);

  part_kernel<<<Bc * NEc, 256, 0, stream>>>(ee, Wh, hpart);
  part_kernel<<<Bc * NEc, 256, 0, stream>>>(ee, Wt, tpart);

  rs_gemm_kernel<<<(Bc * 36 * 48) / 8, 256, 0, stream>>>(htA, seqT, rs);

  extract_gemm_kernel<<<(72 * 48) / 8, 256, 0, stream>>>(rs, WhT, hpart, bh, hsE, 1);
  extract_gemm_kernel<<<(72 * 48) / 8, 256, 0, stream>>>(rs, WtT, tpart, bt, tsE, 0);

  bilinear_proj_kernel<<<dim3(ROWS / 32, Hc / 256, 1), 256, 0, stream>>>(hsE, tsE,
                                                                         WpT, feat);

  classifier_kernel<<<ROWS, 128, 0, stream>>>(feat, Wc, bc, out);
}
